// Segmenter_51178830299223
// MI455X (gfx1250) — compile-verified, matrix operand order checked by reading, not run on hardware
//
#include <hip/hip_runtime.h>

typedef __bf16 bf16;
typedef __attribute__((ext_vector_type(16))) __bf16 v16bf;
typedef __attribute__((ext_vector_type(8)))  float  v8f;
typedef __attribute__((ext_vector_type(4)))  unsigned int u32x4;
typedef __attribute__((ext_vector_type(8)))  int i32x8;
typedef __attribute__((ext_vector_type(4)))  int i32x4;

#define NTOK  8192   // bs * n tokens
#define DFEAT 768
#define KPSI  512
#define KHALF 256
#define HW    1024   // 32x32 token grid per image
#define BS    8
#define KFEAT 32
#define KPIX  10

// ---------------------------------------------------------------------------
// Row-normalize features, emit bf16
// ---------------------------------------------------------------------------
__global__ __launch_bounds__(256) void normalize_rows_kernel(
    const float* __restrict__ f, bf16* __restrict__ out) {
  int row = blockIdx.x;
  const float* r = f + (size_t)row * DFEAT;
  __shared__ float red[256];
  float s = 0.f;
  for (int j = threadIdx.x; j < DFEAT; j += 256) { float v = r[j]; s += v * v; }
  red[threadIdx.x] = s; __syncthreads();
  for (int st = 128; st > 0; st >>= 1) {
    if (threadIdx.x < st) red[threadIdx.x] += red[threadIdx.x + st];
    __syncthreads();
  }
  float inv = rsqrtf(red[0]);
  for (int j = threadIdx.x; j < DFEAT; j += 256)
    out[(size_t)row * DFEAT + j] = (bf16)(r[j] * inv);
}

// ---------------------------------------------------------------------------
// TDM: issue a 2D tensor DMA of a [tile_d1 rows x tile_d0 elems] bf16 tile
// (row stride = stride0 elements) from global into LDS at lds_off.
// D# layout per CDNA5 ISA 8.3/8.4.
// ---------------------------------------------------------------------------
#if __has_builtin(__builtin_amdgcn_tensor_load_to_lds)
#define HAVE_TDM 1
__device__ __forceinline__ void tdm_load_tile_b16(
    unsigned lds_off, unsigned long long gaddr,
    unsigned tensor_d0, unsigned tensor_d1, unsigned long long stride0,
    unsigned tile_d0, unsigned tile_d1) {
  u32x4 g0;
  g0[0] = 1u;                                        // count=1 (valid user D#)
  g0[1] = lds_off;                                   // lds_addr (bytes)
  g0[2] = (unsigned)(gaddr & 0xFFFFFFFFu);           // global_addr[31:0]
  g0[3] = (unsigned)((gaddr >> 32) & 0x1FFFFFFu)     // global_addr[56:32]
        | (2u << 30);                                // type = 2 ("image")
  i32x8 g1;
  g1[0] = (int)(1u << 16);                           // data_size=1 -> 2 bytes
  g1[1] = (int)((tensor_d0 & 0xFFFFu) << 16);        // tensor_dim0[15:0]
  g1[2] = (int)(((tensor_d0 >> 16) & 0xFFFFu) |
                ((tensor_d1 & 0xFFFFu) << 16));      // dim0 hi | dim1 lo
  g1[3] = (int)(((tensor_d1 >> 16) & 0xFFFFu) |
                ((tile_d0 & 0xFFFFu) << 16));        // dim1 hi | tile_dim0
  g1[4] = (int)(tile_d1 & 0xFFFFu);                  // tile_dim1, tile_dim2=0
  g1[5] = (int)(unsigned)(stride0 & 0xFFFFFFFFu);    // tensor_dim0_stride lo
  g1[6] = (int)(unsigned)((stride0 >> 32) & 0xFFFFu);// stride hi | dim1_stride=0
  g1[7] = 0;
  i32x4 z4 = {0, 0, 0, 0};
#if __clang_major__ >= 23
  i32x8 z8 = {0, 0, 0, 0, 0, 0, 0, 0};
  __builtin_amdgcn_tensor_load_to_lds(g0, g1, z4, z4, z8, 0);
#else
  __builtin_amdgcn_tensor_load_to_lds(g0, g1, z4, z4, 0);
#endif
}
#else
#define HAVE_TDM 0
#endif

// ---------------------------------------------------------------------------
// NT GEMM: C[M,N] = A[M,K] * B[N,K]^T  (bf16 in, f32 out)
// Workgroup = 8 waves, computes a 128x16 C block (wave w -> rows tm0+16w).
// B tile (16 rows x 128 K) is DMA'd into LDS by the TDM, double-buffered so
// the DMA of chunk c+1 overlaps the 4 v_wmma_f32_16x16x32_bf16 of chunk c.
// Requires M%128==0, N%16==0, K%128==0, lda/ldb multiples of 8 elements.
// Fragment layouts per CDNA5 ISA 7.12.2 (16-bit A/B striping, f32 C/D).
// ---------------------------------------------------------------------------
__global__ __launch_bounds__(256) void gemm_nt_bf16_tdm(
    const bf16* __restrict__ A, const bf16* __restrict__ B,
    float* __restrict__ C, int N, int K, int lda, int ldb, int ldc) {
  __shared__ __align__(16) bf16 bbuf[2][16 * 128];
  int tn  = blockIdx.x * 16;
  int tm0 = blockIdx.y * 128;
  int tid = threadIdx.x;
  int wave = tid >> 5, lane = tid & 31;
  int r16 = lane & 15, hi = lane >> 4;
  const bf16* arow = A + (size_t)(tm0 + wave * 16 + r16) * lda;
  int nchunk = K >> 7;   // K / 128

#if HAVE_TDM
  unsigned long long bbase =
      (unsigned long long)(size_t)B + 2ull * ((size_t)tn * (size_t)ldb);
  unsigned ldsbase = (unsigned)(size_t)&bbuf[0][0];  // flat addr[31:0] == LDS offset
  if (tid == 0)
    tdm_load_tile_b16(ldsbase, bbase, (unsigned)K, (unsigned)N,
                      (unsigned long long)ldb, 128u, 16u);
#endif

  v8f acc = {};
  union Frag { v16bf v; uint4 u[2]; };

  for (int c = 0; c < nchunk; ++c) {
    int kg = c << 7;
#if HAVE_TDM
    if (tid == 0) __builtin_amdgcn_s_wait_tensorcnt(0);  // buf[c&1] complete
    __syncthreads();                                     // publish to all waves
    if (tid == 0 && c + 1 < nchunk)
      tdm_load_tile_b16(ldsbase + (unsigned)(((c + 1) & 1) * (16 * 128 * 2)),
                        bbase + 2ull * (unsigned long long)(kg + 128),
                        (unsigned)K, (unsigned)N, (unsigned long long)ldb,
                        128u, 16u);
#else
    // Fallback: cooperative synchronous copy (256 thr x 16B = 4KB tile)
    {
      int row = tid >> 4, seg = tid & 15;
      *(uint4*)(&bbuf[c & 1][row * 128 + seg * 8]) =
          *(const uint4*)(B + (size_t)(tn + row) * ldb + kg + seg * 8);
    }
    __syncthreads();
#endif
    const bf16* bl = bbuf[c & 1];
#pragma unroll
    for (int kk = 0; kk < 4; ++kk) {
      Frag a, b;
      a.u[0] = *(const uint4*)(arow + kg + kk * 32 + 8 * hi);
      a.u[1] = *(const uint4*)(arow + kg + kk * 32 + 16 + 8 * hi);
      b.u[0] = *(const uint4*)(bl + r16 * 128 + kk * 32 + 8 * hi);
      b.u[1] = *(const uint4*)(bl + r16 * 128 + kk * 32 + 16 + 8 * hi);
      acc = __builtin_amdgcn_wmma_f32_16x16x32_bf16(
          false, a.v, false, b.v, (short)0, acc, false, false);
    }
    __syncthreads();   // all waves done reading buf[c&1] before it is refilled
  }

  float* crow = C + (size_t)(tm0 + wave * 16 + 8 * hi) * ldc + tn + r16;
#pragma unroll
  for (int r = 0; r < 8; ++r) crow[(size_t)r * ldc] = acc[r];
}

// ---------------------------------------------------------------------------
// clamp(min=0), diag=0, keep only top-32 per row (in place on dense A)
// ---------------------------------------------------------------------------
__global__ __launch_bounds__(256) void feat_topk_kernel(float* __restrict__ A) {
  int i = blockIdx.x;
  __shared__ float srow[NTOK];
  __shared__ float rv[256];
  __shared__ int   ri[256];
  __shared__ int   selIdx[KFEAT];
  __shared__ float selVal[KFEAT];
  float* row = A + (size_t)i * NTOK;
  for (int j = threadIdx.x; j < NTOK; j += 256) {
    float v = row[j];
    v = v < 0.f ? 0.f : v;
    if (j == i) v = 0.f;
    srow[j] = v;
  }
  __syncthreads();
  for (int t = 0; t < KFEAT; ++t) {
    float bv = -1.f; int bi = 0;
    for (int j = threadIdx.x; j < NTOK; j += 256) {
      float v = srow[j];
      if (v > bv) { bv = v; bi = j; }
    }
    rv[threadIdx.x] = bv; ri[threadIdx.x] = bi;
    __syncthreads();
    for (int st = 128; st > 0; st >>= 1) {
      if (threadIdx.x < st) {
        if (rv[threadIdx.x + st] > rv[threadIdx.x] ||
            (rv[threadIdx.x + st] == rv[threadIdx.x] &&
             ri[threadIdx.x + st] < ri[threadIdx.x])) {
          rv[threadIdx.x] = rv[threadIdx.x + st];
          ri[threadIdx.x] = ri[threadIdx.x + st];
        }
      }
      __syncthreads();
    }
    if (threadIdx.x == 0) {
      selIdx[t] = ri[0]; selVal[t] = rv[0];
      srow[ri[0]] = -2.f;
    }
    __syncthreads();
  }
  for (int j = threadIdx.x; j < NTOK; j += 256) row[j] = 0.f;
  __syncthreads();
  if (threadIdx.x < KFEAT) row[selIdx[threadIdx.x]] = selVal[threadIdx.x];
}

// ---------------------------------------------------------------------------
// A <- 0.5*(A + A^T) in place
// ---------------------------------------------------------------------------
__global__ __launch_bounds__(256) void symmetrize_kernel(float* __restrict__ A) {
  int j = blockIdx.x * 16 + (threadIdx.x & 15);
  int i = blockIdx.y * 16 + (threadIdx.x >> 4);
  if (j > i) {
    float v = 0.5f * (A[(size_t)i * NTOK + j] + A[(size_t)j * NTOK + i]);
    A[(size_t)i * NTOK + j] = v;
    A[(size_t)j * NTOK + i] = v;
  }
}

__global__ __launch_bounds__(256) void rowsum_rsqrt_kernel(
    const float* __restrict__ G, float* __restrict__ D, int rowlen) {
  int i = blockIdx.x;
  __shared__ float red[256];
  const float* r = G + (size_t)i * rowlen;
  float s = 0.f;
  for (int j = threadIdx.x; j < rowlen; j += 256) s += r[j];
  red[threadIdx.x] = s; __syncthreads();
  for (int st = 128; st > 0; st >>= 1) {
    if (threadIdx.x < st) red[threadIdx.x] += red[threadIdx.x + st];
    __syncthreads();
  }
  if (threadIdx.x == 0) D[i] = rsqrtf(red[0]);
}

// Gb[i][j] = bf16(G[i][j] * Dr[i] * Dc[blockbase + j]); rowlen = block width
__global__ __launch_bounds__(256) void scale_convert_kernel(
    const float* __restrict__ G, bf16* __restrict__ Gb,
    const float* __restrict__ Dr, const float* __restrict__ Dc,
    int rowlen, size_t total) {
  size_t t = (size_t)blockIdx.x * 256 + threadIdx.x;
  if (t >= total) return;
  size_t i = t / rowlen, j = t % rowlen;
  size_t col = (i / rowlen) * rowlen + j;
  Gb[t] = (bf16)(G[t] * Dr[i] * Dc[col]);
}

__global__ void zero_kernel(float* __restrict__ p, size_t n) {
  size_t t = (size_t)blockIdx.x * 256 + threadIdx.x;
  if (t < n) p[t] = 0.f;
}

// ---------------------------------------------------------------------------
// Pixel kNN: one block per (image b, pixel i)
// ---------------------------------------------------------------------------
__global__ __launch_bounds__(256) void pixel_knn_kernel(
    const float* __restrict__ im, float* __restrict__ Ap, float dw) {
  int b = blockIdx.x >> 10;
  int i = blockIdx.x & (HW - 1);
  __shared__ float d2[HW];
  __shared__ float rv[256];
  __shared__ int   ri[256];
  const float* imb = im + (size_t)b * 3 * HW;
  float zi0 = (imb[i] + 1.f) * 0.5f;
  float zi1 = (imb[HW + i] + 1.f) * 0.5f;
  float zi2 = (imb[2 * HW + i] + 1.f) * 0.5f;
  float xi = (float)(i & 31) * (1.f / 31.f) * dw;
  float yi = (float)(i >> 5) * (1.f / 31.f) * dw;
  for (int j = threadIdx.x; j < HW; j += 256) {
    float a0 = (imb[j] + 1.f) * 0.5f - zi0;
    float a1 = (imb[HW + j] + 1.f) * 0.5f - zi1;
    float a2 = (imb[2 * HW + j] + 1.f) * 0.5f - zi2;
    float a3 = (float)(j & 31) * (1.f / 31.f) * dw - xi;
    float a4 = (float)(j >> 5) * (1.f / 31.f) * dw - yi;
    float d = a0 * a0 + a1 * a1 + a2 * a2 + a3 * a3 + a4 * a4;
    d2[j] = (j == i) ? 3.0e38f : d;
  }
  __syncthreads();
  float* apb = Ap + (size_t)b * HW * HW;
  for (int t = 0; t < KPIX; ++t) {
    float bv = 3.4e38f; int bi = 0;
    for (int j = threadIdx.x; j < HW; j += 256) {
      float v = d2[j];
      if (v < bv) { bv = v; bi = j; }
    }
    rv[threadIdx.x] = bv; ri[threadIdx.x] = bi;
    __syncthreads();
    for (int st = 128; st > 0; st >>= 1) {
      if (threadIdx.x < st) {
        if (rv[threadIdx.x + st] < rv[threadIdx.x] ||
            (rv[threadIdx.x + st] == rv[threadIdx.x] &&
             ri[threadIdx.x + st] < ri[threadIdx.x])) {
          rv[threadIdx.x] = rv[threadIdx.x + st];
          ri[threadIdx.x] = ri[threadIdx.x + st];
        }
      }
      __syncthreads();
    }
    if (threadIdx.x == 0) {
      int j = ri[0];
      apb[(size_t)i * HW + j] = 1.f;
      apb[(size_t)j * HW + i] = 1.f;
      d2[j] = 3.4e38f;
    }
    __syncthreads();
  }
}

// ---------------------------------------------------------------------------
__global__ void psit_scale_kernel(const float* __restrict__ Psi,
                                  bf16* __restrict__ Ptb) {
  size_t t = (size_t)blockIdx.x * 256 + threadIdx.x;
  if (t >= (size_t)NTOK * KPSI) return;
  size_t n = t / KPSI, c = t % KPSI;
  const float s = 0.0349385580f;   // sqrt(10 / 8192)
  Ptb[c * (size_t)NTOK + n] = (bf16)(Psi[t] * s);
}

__global__ void gp_transpose_kernel(const float* __restrict__ GP,
                                    bf16* __restrict__ GPt) {
  size_t t = (size_t)blockIdx.x * 256 + threadIdx.x;
  if (t >= (size_t)NTOK * KPSI) return;
  size_t n = t / KPSI, c = t % KPSI;
  GPt[c * (size_t)NTOK + n] = (bf16)GP[t];
}

// ---------------------------------------------------------------------------
// loss = -(tr R1 + tr R2)/K ; reg = alpha*(triu(R1^2)+triu(R2^2))/K
// ---------------------------------------------------------------------------
__global__ __launch_bounds__(256) void finalize_kernel(
    const float* __restrict__ R, float* __restrict__ out) {
  __shared__ float redT[256], redU[256];
  int i = threadIdx.x;
  float tr = 0.f, tu = 0.f;
  for (int m = 0; m < 2; ++m) {
    const float* Rm = R + (size_t)m * KHALF * KHALF;
    tr += Rm[(size_t)i * KHALF + i];
    for (int j = i + 1; j < KHALF; ++j) {
      float v = Rm[(size_t)i * KHALF + j];
      tu += v * v;
    }
  }
  redT[i] = tr; redU[i] = tu; __syncthreads();
  for (int st = 128; st > 0; st >>= 1) {
    if (i < st) { redT[i] += redT[i + st]; redU[i] += redU[i + st]; }
    __syncthreads();
  }
  if (i == 0) {
    out[0] = -redT[0] / (float)KPSI;
    out[1] = 0.05f * redU[0] / (float)KPSI;
  }
}

// ---------------------------------------------------------------------------
extern "C" void kernel_launch(void* const* d_in, const int* in_sizes, int n_in,
                              void* d_out, int out_size, void* d_ws, size_t ws_size,
                              hipStream_t stream) {
  (void)in_sizes; (void)n_in; (void)out_size; (void)ws_size;
  const float* hf  = (const float*)d_in[0];  // [8,1024,768]
  const float* Psi = (const float*)d_in[1];  // [8192,512]
  const float* im  = (const float*)d_in[2];  // [8,3,32,32]
  float* out = (float*)d_out;

  char* ws = (char*)d_ws;
  size_t off = 0;
  auto alloc = [&](size_t bytes) -> char* {
    char* p = ws + off;
    off = (off + bytes + 255) & ~(size_t)255;
    return p;
  };
  bf16*  fnb  = (bf16*) alloc((size_t)NTOK * DFEAT * 2);   // 12.6 MB
  float* Abuf = (float*)alloc((size_t)NTOK * NTOK * 4);    // 268 MB (A -> G1)
  bf16*  G1b  = (bf16*) alloc((size_t)NTOK * NTOK * 2);    // 134 MB
  float* D1   = (float*)alloc((size_t)NTOK * 4);
  float* Ap   = (float*)alloc((size_t)BS * HW * HW * 4);   // 33.5 MB
  float* D2   = (float*)alloc((size_t)NTOK * 4);
  bf16*  G2b  = (bf16*) alloc((size_t)BS * HW * HW * 2);   // 16.8 MB
  bf16*  Ptb  = (bf16*) alloc((size_t)KPSI * NTOK * 2);    // 8.4 MB
  float* GP   = (float*)alloc((size_t)NTOK * KPSI * 4);    // 16.8 MB
  bf16*  GPtb = (bf16*) alloc((size_t)KPSI * NTOK * 2);    // 8.4 MB
  float* Rbuf = (float*)alloc((size_t)2 * KHALF * KHALF * 4);

  // --- feature graph G1 ---
  normalize_rows_kernel<<<NTOK, 256, 0, stream>>>(hf, fnb);
  gemm_nt_bf16_tdm<<<dim3(NTOK / 16, NTOK / 128), 256, 0, stream>>>(
      fnb, fnb, Abuf, NTOK, DFEAT, DFEAT, DFEAT, NTOK);
  feat_topk_kernel<<<NTOK, 256, 0, stream>>>(Abuf);
  symmetrize_kernel<<<dim3(NTOK / 16, NTOK / 16), 256, 0, stream>>>(Abuf);
  rowsum_rsqrt_kernel<<<NTOK, 256, 0, stream>>>(Abuf, D1, NTOK);
  {
    size_t total = (size_t)NTOK * NTOK;
    scale_convert_kernel<<<(unsigned)((total + 255) / 256), 256, 0, stream>>>(
        Abuf, G1b, D1, D1, NTOK, total);
  }

  // --- pixel graph G2 (block diagonal) ---
  {
    size_t n = (size_t)BS * HW * HW;
    zero_kernel<<<(unsigned)((n + 255) / 256), 256, 0, stream>>>(Ap, n);
  }
  pixel_knn_kernel<<<BS * HW, 256, 0, stream>>>(im, Ap, 2.0f);
  pixel_knn_kernel<<<BS * HW, 256, 0, stream>>>(im, Ap, 0.1f);
  rowsum_rsqrt_kernel<<<NTOK, 256, 0, stream>>>(Ap, D2, HW);
  {
    size_t total = (size_t)BS * HW * HW;
    scale_convert_kernel<<<(unsigned)((total + 255) / 256), 256, 0, stream>>>(
        Ap, G2b, D2, D2, HW, total);
  }

  // --- Psi scaled + transposed to bf16 ---
  {
    size_t n = (size_t)NTOK * KPSI;
    psit_scale_kernel<<<(unsigned)((n + 255) / 256), 256, 0, stream>>>(Psi, Ptb);
  }

  // --- GP1 = G1 @ P1  (cols 0..255 of GP) ---
  gemm_nt_bf16_tdm<<<dim3(KHALF / 16, NTOK / 128), 256, 0, stream>>>(
      G1b, Ptb, GP, KHALF, NTOK, NTOK, NTOK, KPSI);
  // --- GP2 = blockdiag(G2) @ P2  (cols 256..511 of GP), per image block ---
  for (int b = 0; b < BS; ++b) {
    gemm_nt_bf16_tdm<<<dim3(KHALF / 16, HW / 128), 256, 0, stream>>>(
        G2b + (size_t)b * HW * HW,
        Ptb + (size_t)KHALF * NTOK + (size_t)b * HW,
        GP + (size_t)b * HW * KPSI + KHALF,
        KHALF, HW, HW, NTOK, KPSI);
  }

  // --- R = P^T @ GP (both halves), via GP^T so the GEMM stays NT ---
  {
    size_t n = (size_t)NTOK * KPSI;
    gp_transpose_kernel<<<(unsigned)((n + 255) / 256), 256, 0, stream>>>(GP, GPtb);
  }
  gemm_nt_bf16_tdm<<<dim3(KHALF / 16, KHALF / 128), 256, 0, stream>>>(
      Ptb, GPtb, Rbuf, KHALF, NTOK, NTOK, NTOK, KHALF);
  gemm_nt_bf16_tdm<<<dim3(KHALF / 16, KHALF / 128), 256, 0, stream>>>(
      Ptb + (size_t)KHALF * NTOK, GPtb + (size_t)KHALF * NTOK,
      Rbuf + (size_t)KHALF * KHALF, KHALF, NTOK, NTOK, NTOK, KHALF);

  finalize_kernel<<<1, 256, 0, stream>>>(Rbuf, out);
}
